// GIN_KAN_68539088110036
// MI455X (gfx1250) — compile-verified, matrix operand order checked by reading, not run on hardware
//
#include <hip/hip_runtime.h>
#include <hip/hip_bf16.h>

// ---------------------------------------------------------------------------
// GIN-KAN forward for gfx1250. All matrix work goes through
// v_wmma_f32_16x16x32_bf16 (bf16 A/B, f32 accum). KAN layers are folded into
// a single GEMM with an augmented input [silu(x_i), B-spline bases(x_i)].
// GEMM kernels use 2x2 register tiles per wave (4 WMMA chains) and 32-row
// LDS staging to cut L2 weight traffic in half.
// ---------------------------------------------------------------------------

typedef __attribute__((ext_vector_type(16))) __bf16 v16bf;
typedef __attribute__((ext_vector_type(8)))  float  v8f;

constexpr int Nn  = 8192;
constexpr int Hh  = 128;
constexpr int Ee  = 65536;
constexpr int Gg  = 16;
constexpr int OUTd = 64;
constexpr int ECk = 8192;      // edge chunk
constexpr int NCHUNK = Ee / ECk;

// ---------------- small math helpers ----------------

__device__ __forceinline__ float siluf(float x) { return x / (1.f + __expf(-x)); }
__device__ __forceinline__ float sigmf(float x) { return 1.f / (1.f + __expf(-x)); }

// Cubic B-spline bases on efficient-kan grid: grid_size=5, order=3,
// knots t(j) = 0.4*j - 2.2, j=0..11  ->  8 bases.
__device__ __forceinline__ void bspline8(float x, float* out) {
  float b[11];
#pragma unroll
  for (int j = 0; j < 11; ++j) {
    float t0 = 0.4f * j - 2.2f;
    float t1 = t0 + 0.4f;
    b[j] = (x >= t0 && x < t1) ? 1.f : 0.f;
  }
#pragma unroll
  for (int ord = 1; ord <= 3; ++ord) {
    float inv = 1.f / (0.4f * (float)ord);
#pragma unroll
    for (int j = 0; j < 10; ++j) {
      if (j < 11 - ord) {
        float t0 = 0.4f * j - 2.2f;
        float tn = 0.4f * (float)(j + ord + 1) - 2.2f;
        b[j] = (x - t0) * inv * b[j] + (tn - x) * inv * b[j + 1];
      }
    }
  }
#pragma unroll
  for (int g = 0; g < 8; ++g) out[g] = b[g];
}

// load 16 bf16 per lane following the documented 16-bit A/B fragment layout:
// p points at rowBase + k0 + koff; elements {p[0..7], p[16..23]}
__device__ __forceinline__ v16bf load_frag16(const __bf16* p) {
  v16bf r;
#pragma unroll
  for (int j = 0; j < 8; ++j) { r[j] = p[j]; r[j + 8] = p[j + 16]; }
  return r;
}

// ---------------- elementwise / utility kernels ----------------

__global__ void zero_f32(float* p, int n) {
  int i = blockIdx.x * blockDim.x + threadIdx.x;
  if (i < n) p[i] = 0.f;
}

__global__ void copy_f32(float* dst, const float* src, int n) {
  int i = blockIdx.x * blockDim.x + threadIdx.x;
  if (i < n) dst[i] = src[i];
}

__global__ void f32_to_bf16(const float* in, __bf16* out, int n) {
  int i = blockIdx.x * blockDim.x + threadIdx.x;
  if (i < n) out[i] = (__bf16)in[i];
}

__global__ void gelu_inplace(float* p, int n) {
  int i = blockIdx.x * blockDim.x + threadIdx.x;
  if (i < n) {
    float x = p[i];
    p[i] = 0.5f * x * (1.f + erff(x * 0.70710678118654752f));
  }
}

// out[src[e]] += (alpha?alpha[e]:1) * h[dst[e]]   (row-wise, H=128)
__global__ void scatter_add_rows(float* out, const float* h, const int* src,
                                 const int* dstI, const float* alpha, int total) {
  int i = blockIdx.x * blockDim.x + threadIdx.x;
  if (i >= total) return;
  int e = i >> 7;
  int f = i & 127;
  float a = alpha ? alpha[e] : 1.f;
  atomicAdd(&out[src[e] * Hh + f], a * h[dstI[e] * Hh + f]);
}

__global__ void pool_graph(float* gpool, const float* h, const int* gid, int total) {
  int i = blockIdx.x * blockDim.x + threadIdx.x;
  if (i >= total) return;
  int n = i >> 7;
  int f = i & 127;
  atomicAdd(&gpool[gid[n] * Hh + f], h[i]);
}

// ---------------- KAN weight prep: Wa[o, 9i+0]=base_w, 9i+1+g = spline*scaler --

__global__ void prep_kan_w(const float* bw, const float* sc, const float* sw,
                           __bf16* Wa, int Out, int In) {
  int t = blockIdx.x * blockDim.x + threadIdx.x;
  if (t >= Out * In) return;
  int o = t / In, i = t - o * In;
  __bf16* d = Wa + (size_t)o * In * 9 + (size_t)i * 9;
  d[0] = (__bf16)bw[t];
  float s = sc[t];
#pragma unroll
  for (int g = 0; g < 8; ++g) d[1 + g] = (__bf16)(sw[t * 8 + g] * s);
}

// ---------------- fused KAN GEMM: C[M,Out] = Aug(X) @ Wa^T -------------------
// X cols: i<In1 from X1 (optionally row-gathered via idx1), else from X2.
// Phase 1 builds the augmented bf16 tile (32 rows x In*9) in LDS.
// Phase 2: wave w computes a 32x32 output tile (2x2 WMMA register tile) at
// output cols [32w, 32w+32). blockDim = 32 * (Out/32).
__global__ void kan_gemm(const float* __restrict__ X1, const float* __restrict__ X2,
                         const int* __restrict__ idx1, const __bf16* __restrict__ Wa,
                         float* __restrict__ C, int In1, int In, int Out) {
  extern __shared__ char smem_raw[];
  __bf16* aug = reinterpret_cast<__bf16*>(smem_raw);   // [32][In*9]
  const int KA = In * 9;
  const int row0 = blockIdx.x * 32;

  // phase 1: augmentation (32 rows)
  for (int t = threadIdx.x; t < 32 * In; t += blockDim.x) {
    int r = t / In, i = t - r * In;
    int grow = row0 + r;
    float x;
    if (i < In1) {
      int rr = idx1 ? idx1[grow] : grow;
      x = X1[(size_t)rr * In1 + i];
    } else {
      x = X2[(size_t)grow * (In - In1) + (i - In1)];
    }
    float bs[8];
    bspline8(x, bs);
    __bf16* d = aug + (size_t)r * KA + (size_t)i * 9;
    d[0] = (__bf16)siluf(x);
#pragma unroll
    for (int g = 0; g < 8; ++g) d[1 + g] = (__bf16)bs[g];
  }
  __syncthreads();

  // phase 2: 2x2 register tile of WMMAs over K = In*9
  const int wave = threadIdx.x >> 5;
  const int lane = threadIdx.x & 31;
  const int col0 = wave * 32;
  const int koff = (lane >> 4) * 8;

  const __bf16* arow0 = aug + (size_t)(lane & 15) * KA + koff;          // rows 0..15
  const __bf16* arow1 = aug + (size_t)((lane & 15) + 16) * KA + koff;   // rows 16..31
  const __bf16* wrow0 = Wa + (size_t)(col0 + (lane & 15)) * KA + koff;
  const __bf16* wrow1 = Wa + (size_t)(col0 + 16 + (lane & 15)) * KA + koff;

  v8f acc00 = {}, acc01 = {}, acc10 = {}, acc11 = {};
#pragma unroll 2
  for (int k0 = 0; k0 < KA; k0 += 32) {
    __builtin_prefetch(wrow0 + k0 + 128, 0, 1);
    __builtin_prefetch(wrow1 + k0 + 128, 0, 1);
    v16bf a0 = load_frag16(arow0 + k0);
    v16bf a1 = load_frag16(arow1 + k0);
    v16bf b0 = load_frag16(wrow0 + k0);
    v16bf b1 = load_frag16(wrow1 + k0);
    acc00 = __builtin_amdgcn_wmma_f32_16x16x32_bf16(false, a0, false, b0, (short)0, acc00, false, false);
    acc01 = __builtin_amdgcn_wmma_f32_16x16x32_bf16(false, a0, false, b1, (short)0, acc01, false, false);
    acc10 = __builtin_amdgcn_wmma_f32_16x16x32_bf16(false, a1, false, b0, (short)0, acc10, false, false);
    acc11 = __builtin_amdgcn_wmma_f32_16x16x32_bf16(false, a1, false, b1, (short)0, acc11, false, false);
  }
  const int col   = lane & 15;
  const int rbase = (lane >> 4) * 8;
#pragma unroll
  for (int r = 0; r < 8; ++r) {
    float* c0 = C + (size_t)(row0 + rbase + r) * Out + col0 + col;
    float* c1 = C + (size_t)(row0 + 16 + rbase + r) * Out + col0 + col;
    c0[0]  = acc00[r];
    c0[16] = acc01[r];
    c1[0]  = acc10[r];
    c1[16] = acc11[r];
  }
}

// ---------------- generic bf16 WMMA GEMM with optional row gather ------------
// C[M,Nout] = A[idx[m], 0..K) @ W[Nout,K]^T (+bias). 32x32 tile per wave.
__global__ void wmma_gemm_gather(const __bf16* __restrict__ A, const int* __restrict__ idx,
                                 const __bf16* __restrict__ W, const float* __restrict__ bias,
                                 float* __restrict__ C, int M, int Nout, int K) {
  const int wave = threadIdx.x >> 5;
  const int lane = threadIdx.x & 31;
  const int ctiles = Nout >> 5;
  const int tile = blockIdx.x * (blockDim.x >> 5) + wave;
  if (tile >= (M >> 5) * ctiles) return;          // whole-wave uniform branch
  const int row0 = (tile / ctiles) * 32;
  const int col0 = (tile % ctiles) * 32;
  const int koff = (lane >> 4) * 8;

  int ar0 = row0 + (lane & 15);
  int ar1 = ar0 + 16;
  if (idx) { ar0 = idx[ar0]; ar1 = idx[ar1]; }
  const __bf16* pa0 = A + (size_t)ar0 * K + koff;
  const __bf16* pa1 = A + (size_t)ar1 * K + koff;
  const __bf16* pw0 = W + (size_t)(col0 + (lane & 15)) * K + koff;
  const __bf16* pw1 = W + (size_t)(col0 + 16 + (lane & 15)) * K + koff;

  v8f acc00 = {}, acc01 = {}, acc10 = {}, acc11 = {};
#pragma unroll 2
  for (int k0 = 0; k0 < K; k0 += 32) {
    v16bf a0 = load_frag16(pa0 + k0);
    v16bf a1 = load_frag16(pa1 + k0);
    v16bf b0 = load_frag16(pw0 + k0);
    v16bf b1 = load_frag16(pw1 + k0);
    acc00 = __builtin_amdgcn_wmma_f32_16x16x32_bf16(false, a0, false, b0, (short)0, acc00, false, false);
    acc01 = __builtin_amdgcn_wmma_f32_16x16x32_bf16(false, a0, false, b1, (short)0, acc01, false, false);
    acc10 = __builtin_amdgcn_wmma_f32_16x16x32_bf16(false, a1, false, b0, (short)0, acc10, false, false);
    acc11 = __builtin_amdgcn_wmma_f32_16x16x32_bf16(false, a1, false, b1, (short)0, acc11, false, false);
  }
  const int col   = lane & 15;
  const int rbase = (lane >> 4) * 8;
  const float bv0 = bias ? bias[col0 + col] : 0.f;
  const float bv1 = bias ? bias[col0 + 16 + col] : 0.f;
#pragma unroll
  for (int r = 0; r < 8; ++r) {
    float* c0 = C + (size_t)(row0 + rbase + r) * Nout + col0 + col;
    float* c1 = C + (size_t)(row0 + 16 + rbase + r) * Nout + col0 + col;
    c0[0]  = acc00[r] + bv0;
    c0[16] = acc01[r] + bv1;
    c1[0]  = acc10[r] + bv0;
    c1[16] = acc11[r] + bv1;
  }
}

// ---------------- BN stats: per-feature sum / sumsq over rows ----------------
__global__ void bn_stats(const float* rep, float* stats, int Nrows) {
  __shared__ float s1[256], s2[256];
  int f = blockIdx.x;
  float a = 0.f, b = 0.f;
  for (int r = threadIdx.x; r < Nrows; r += 256) {
    float v = rep[(size_t)r * Hh + f];
    a += v; b += v * v;
  }
  s1[threadIdx.x] = a; s2[threadIdx.x] = b;
  __syncthreads();
  for (int st = 128; st > 0; st >>= 1) {
    if (threadIdx.x < st) { s1[threadIdx.x] += s1[threadIdx.x + st];
                            s2[threadIdx.x] += s2[threadIdx.x + st]; }
    __syncthreads();
  }
  if (threadIdx.x == 0) { stats[f] = s1[0]; stats[Hh + f] = s2[0]; }
}

// ---------------- BN affine + ReLU + gated residual + LayerNorm --------------
__global__ void postbn_ln(const float* rep, const float* stats,
                          const float* bn_g, const float* bn_b,
                          const float* resid, const float* gate_p,
                          const float* ln_g, const float* ln_b,
                          float* hnext, __bf16* hnb, int Nrows) {
  __shared__ float s1[128], s2[128];
  int r = blockIdx.x, f = threadIdx.x;
  float invN = 1.f / (float)Nrows;
  float mu  = stats[f] * invN;
  float var = stats[Hh + f] * invN - mu * mu;
  float v = (rep[(size_t)r * Hh + f] - mu) * rsqrtf(var + 1e-5f) * bn_g[f] + bn_b[f];
  v = fmaxf(v, 0.f);
  float g = sigmf(gate_p[0]);
  float hn = g * v + (1.f - g) * resid[(size_t)r * Hh + f];
  s1[f] = hn; s2[f] = hn * hn;
  __syncthreads();
  for (int st = 64; st > 0; st >>= 1) {
    if (f < st) { s1[f] += s1[f + st]; s2[f] += s2[f + st]; }
    __syncthreads();
  }
  float m  = s1[0] * (1.f / 128.f);
  float vv = s2[0] * (1.f / 128.f) - m * m;
  float o = (hn - m) * rsqrtf(vv + 1e-5f) * ln_g[f] + ln_b[f];
  hnext[(size_t)r * Hh + f] = o;
  hnb[(size_t)r * Hh + f] = (__bf16)o;
}

// ---------------- attention: softmax over 4 heads, weight V ------------------
__global__ void attn_combine(const float* q, const float* k, const float* v,
                             const float* qb, const float* kb, const float* vb,
                             const float* pe, float* att) {
  __shared__ float red[128];
  __shared__ float wgt[4];
  int e = blockIdx.x, f = threadIdx.x;
  size_t o = (size_t)e * 128 + f;
  float qv = q[o] + qb[f] + pe[f];
  float kv = k[o] + kb[f] + pe[f];
  float vv = v[o] + vb[f];
  red[f] = qv * kv;
  __syncthreads();
  if (f < 4) {
    float s = 0.f;
#pragma unroll
    for (int d = 0; d < 32; ++d) s += red[f * 32 + d];
    wgt[f] = s * 0.17677669529663688f;   // 1/sqrt(32)
  }
  __syncthreads();
  if (f == 0) {
    float m = fmaxf(fmaxf(wgt[0], wgt[1]), fmaxf(wgt[2], wgt[3]));
    float e0 = __expf(wgt[0] - m), e1 = __expf(wgt[1] - m);
    float e2 = __expf(wgt[2] - m), e3 = __expf(wgt[3] - m);
    float inv = 1.f / (e0 + e1 + e2 + e3);
    wgt[0] = e0 * inv; wgt[1] = e1 * inv; wgt[2] = e2 * inv; wgt[3] = e3 * inv;
  }
  __syncthreads();
  att[o] = wgt[f >> 5] * vv;
}

// ---------------- KAN2 (Out=1): augmented dot + sigmoid ----------------------
__global__ void kan2_alpha(const float* z, const float* bw, const float* sc,
                           const float* sw, float* alpha) {
  __shared__ float red[128];
  int e = blockIdx.x, f = threadIdx.x;
  float x = z[(size_t)e * 128 + f];
  float bs[8];
  bspline8(x, bs);
  float acc = siluf(x) * bw[f];
  float scl = sc[f];
#pragma unroll
  for (int g = 0; g < 8; ++g) acc += bs[g] * sw[f * 8 + g] * scl;
  red[f] = acc;
  __syncthreads();
  for (int st = 64; st > 0; st >>= 1) {
    if (f < st) red[f] += red[f + st];
    __syncthreads();
  }
  if (f == 0) alpha[e] = sigmf(red[0]);
}

// ---------------- prediction: out[g,o] += gpool[g] . pred_w[o] + pred_b[o] ---
__global__ void pred_accum(const float* gpool, const float* pw, const float* pb,
                           float* out) {
  int g = blockIdx.x, o = threadIdx.x;
  float acc = pb[o];
  const float* gp = gpool + (size_t)g * Hh;
  const float* w  = pw + (size_t)o * Hh;
  for (int f = 0; f < Hh; ++f) acc += gp[f] * w[f];
  out[g * OUTd + o] += acc;
}

// ---------------------------------------------------------------------------

static inline char* wsal(char*& p, size_t bytes) {
  char* r = p;
  p += (bytes + 255) & ~(size_t)255;
  return r;
}

extern "C" void kernel_launch(void* const* d_in, const int* in_sizes, int n_in,
                              void* d_out, int out_size, void* d_ws, size_t ws_size,
                              hipStream_t stream) {
  (void)in_sizes; (void)n_in; (void)out_size; (void)ws_size;
  auto F = [&](int i) { return (const float*)d_in[i]; };

  const float* x  = F(0);
  const int* ei   = (const int*)d_in[1];
  const int* src  = ei;
  const int* dst  = ei + Ee;
  const int* gid  = (const int*)d_in[2];
  // params flatten order (sorted dict keys / list order):
  // layer base B: k_b,k_w, kan1{base_w,scaler,spline_w}, kan2{...}, pos_enc,
  //               q_b,q_w, v_b,v_w, bn_b,bn_g, kan{...}, ln_b,ln_g, res_gate,
  //               [skip_b, skip_w]          (layer0 only)
  // then pred_b (47), pred_w (48)
  const int B0 = 3, B1 = 26;
  const float* pred_b = F(47);
  const float* pred_w = F(48);

  // workspace carve
  char* p = (char*)d_ws;
  float*  h1     = (float*)wsal(p, (size_t)Nn * Hh * 4);
  float*  h2     = (float*)wsal(p, (size_t)Nn * Hh * 4);
  float*  pooled = (float*)wsal(p, (size_t)Nn * Hh * 4);   // later reused as h_next
  float*  rep    = (float*)wsal(p, (size_t)Nn * Hh * 4);
  float*  resid  = (float*)wsal(p, (size_t)Nn * Hh * 4);
  __bf16* hnb    = (__bf16*)wsal(p, (size_t)Nn * Hh * 2);
  __bf16* xbf    = (__bf16*)wsal(p, (size_t)Nn * Hh * 2);
  __bf16* waKan  = (__bf16*)wsal(p, (size_t)128 * 2304 * 2);
  __bf16* wqb    = (__bf16*)wsal(p, (size_t)128 * 128 * 2);
  __bf16* wkb    = (__bf16*)wsal(p, (size_t)128 * 128 * 2);
  __bf16* wvb    = (__bf16*)wsal(p, (size_t)128 * 128 * 2);
  __bf16* wskb   = (__bf16*)wsal(p, (size_t)128 * 128 * 2);
  float*  stats  = (float*)wsal(p, 2 * Hh * 4);
  float*  qbuf   = (float*)wsal(p, (size_t)ECk * Hh * 4);
  float*  kbuf   = (float*)wsal(p, (size_t)ECk * Hh * 4);
  float*  vbuf   = (float*)wsal(p, (size_t)ECk * Hh * 4);
  float*  attb   = (float*)wsal(p, (size_t)ECk * Hh * 4);
  float*  k1out  = (float*)wsal(p, (size_t)ECk * Hh * 4);
  float*  alpha  = (float*)wsal(p, (size_t)Ee * 4);
  float*  gpool  = (float*)wsal(p, (size_t)Gg * Hh * 4);
  float*  out    = (float*)d_out;

  const int NH = Nn * Hh;           // 1,048,576
  const int EH = Ee * Hh;           // 8,388,608

  // x -> bf16 (for layer-0 skip GEMM)
  f32_to_bf16<<<NH / 256, 256, 0, stream>>>(x, xbf, NH);

  for (int l = 0; l < 2; ++l) {
    const float* hin = (l == 0) ? x : h1;
    float* hout = (l == 0) ? h1 : h2;
    const int B = (l == 0) ? B0 : B1;

    // ---- GIN pooling: pooled = self + neighbor sum
    copy_f32<<<NH / 256, 256, 0, stream>>>(pooled, hin, NH);
    scatter_add_rows<<<EH / 256, 256, 0, stream>>>(pooled, hin, src, dst, nullptr, EH);

    // ---- node KAN: rep = Aug(pooled) @ Wa^T   (K = 128*9 = 1152)
    prep_kan_w<<<(128 * 128 + 255) / 256, 256, 0, stream>>>(F(B + 15), F(B + 16), F(B + 17),
                                                            waKan, 128, 128);
    kan_gemm<<<Nn / 32, 128, 32 * 128 * 9 * 2, stream>>>(pooled, nullptr, nullptr,
                                                         waKan, rep, 128, 128, 128);

    // ---- BatchNorm statistics
    bn_stats<<<Hh, 256, 0, stream>>>(rep, stats, Nn);

    // ---- residual branch
    const float* resptr;
    if (l == 0) {
      f32_to_bf16<<<(128 * 128 + 255) / 256, 256, 0, stream>>>(F(B + 22), wskb, 128 * 128);
      wmma_gemm_gather<<<(Nn / 32) * 4 / 8, 256, 0, stream>>>(xbf, nullptr, wskb, F(B + 21),
                                                              resid, Nn, 128, 128);
      resptr = resid;
    } else {
      resptr = h1;
    }

    // ---- BN affine + ReLU + gate + LN -> h_next (pooled buf) + bf16 copy
    postbn_ln<<<Nn, 128, 0, stream>>>(rep, stats, F(B + 14), F(B + 13), resptr,
                                      F(B + 20), F(B + 19), F(B + 18),
                                      pooled, hnb, Nn);

    // ---- attention weights to bf16, KAN1 augmented weights
    f32_to_bf16<<<(128 * 128 + 255) / 256, 256, 0, stream>>>(F(B + 10), wqb, 128 * 128);
    f32_to_bf16<<<(128 * 128 + 255) / 256, 256, 0, stream>>>(F(B + 1),  wkb, 128 * 128);
    f32_to_bf16<<<(128 * 128 + 255) / 256, 256, 0, stream>>>(F(B + 12), wvb, 128 * 128);
    prep_kan_w<<<(128 * 256 + 255) / 256, 256, 0, stream>>>(F(B + 2), F(B + 3), F(B + 4),
                                                            waKan, 128, 256);

    // ---- per-edge attention + edge KAN, chunked to bound workspace
    for (int c = 0; c < NCHUNK; ++c) {
      const int e0 = c * ECk;
      // gather-GEMMs: q from h_next[src], k/v from h_next[dst]
      wmma_gemm_gather<<<(ECk / 32) * 4 / 8, 256, 0, stream>>>(hnb, src + e0, wqb, nullptr,
                                                               qbuf, ECk, 128, 128);
      wmma_gemm_gather<<<(ECk / 32) * 4 / 8, 256, 0, stream>>>(hnb, dst + e0, wkb, nullptr,
                                                               kbuf, ECk, 128, 128);
      wmma_gemm_gather<<<(ECk / 32) * 4 / 8, 256, 0, stream>>>(hnb, dst + e0, wvb, nullptr,
                                                               vbuf, ECk, 128, 128);
      attn_combine<<<ECk, 128, 0, stream>>>(qbuf, kbuf, vbuf, F(B + 9), F(B + 0), F(B + 11),
                                            F(B + 8), attb);
      // KAN1 over z = [h_next[src], attended] : K = 256*9 = 2304
      kan_gemm<<<ECk / 32, 128, 32 * 256 * 9 * 2, stream>>>(pooled, attb, src + e0,
                                                            waKan, k1out, 128, 256, 128);
      gelu_inplace<<<(ECk * 128) / 256, 256, 0, stream>>>(k1out, ECk * 128);
      kan2_alpha<<<ECk, 128, 0, stream>>>(k1out, F(B + 5), F(B + 6), F(B + 7), alpha + e0);
    }

    // ---- h_out = h_next + scatter(alpha * h_next[dst])
    copy_f32<<<NH / 256, 256, 0, stream>>>(hout, pooled, NH);
    scatter_add_rows<<<EH / 256, 256, 0, stream>>>(hout, pooled, src, dst, alpha, EH);
  }

  // ---- prediction head over hidden states {x, h1, h2}
  zero_f32<<<(Gg * OUTd + 255) / 256, 256, 0, stream>>>(out, Gg * OUTd);
  const float* levels[3] = {x, h1, h2};
  for (int l = 0; l < 3; ++l) {
    zero_f32<<<(Gg * Hh + 255) / 256, 256, 0, stream>>>(gpool, Gg * Hh);
    pool_graph<<<NH / 256, 256, 0, stream>>>(gpool, levels[l], gid, NH);
    pred_accum<<<Gg, OUTd, 0, stream>>>(gpool, pred_w + (size_t)l * OUTd * Hh,
                                        pred_b + (size_t)l * OUTd, out);
  }
}
